// Model_83519934038723
// MI455X (gfx1250) — matrix operand order repaired and checked by reading, not compiled, not run on hardware
//
#include <hip/hip_runtime.h>
#include <math.h>

namespace {

constexpr int       kN   = 65536;
constexpr int       kE   = 524288;
constexpr int       kG   = 256;
constexpr int       kH   = 64;
constexpr int       kNF  = 9;
constexpr int       kV   = 128;
constexpr long long kD   = (long long)kN * kH;   // 4,194,304 floats per iterate

typedef __attribute__((ext_vector_type(16))) _Float16 v16h;
typedef __attribute__((ext_vector_type(8)))  float    v8f;

// ---------------------------------------------------------------- fill
__global__ __launch_bounds__(256)
void fill_kernel(float* __restrict__ p, float v, long long n)
{
    long long i = (long long)blockIdx.x * blockDim.x + threadIdx.x;
    long long s = (long long)gridDim.x * blockDim.x;
    for (; i < n; i += s) p[i] = v;
}

// ---------------------------------------------------- atom encoder + relu
__global__ __launch_bounds__(256)
void atom_enc_kernel(const int* __restrict__ data_x,   // [N, 9]
                     const float* __restrict__ emb,    // [9, 128, 64]
                     float* __restrict__ h)            // [N, 64]
{
    int tid  = blockIdx.x * 256 + threadIdx.x;
    int node = tid >> 6;
    int c    = tid & 63;
    float s = 0.f;
#pragma unroll
    for (int col = 0; col < kNF; ++col) {
        int v = data_x[node * kNF + col];
        s += emb[((long long)col * kV + v) * kH + c];
    }
    h[(long long)node * kH + c] = fmaxf(s, 0.f);
}

// -------------------------------------------- Cayley: Wc = (I+S)^-1 (I-S)
__global__ __launch_bounds__(64)
void cayley_kernel(const float* __restrict__ Mw, float* __restrict__ Wc)
{
    __shared__ float aug[64][130];
    int r = threadIdx.x;
    for (int c = 0; c < 64; ++c) {
        float S = 0.5f * (Mw[r * 64 + c] - Mw[c * 64 + r]);
        float id = (r == c) ? 1.f : 0.f;
        aug[r][c]      = id + S;   // I + S
        aug[r][64 + c] = id - S;   // I - S
    }
    __syncthreads();
    for (int p = 0; p < 64; ++p) {
        float f = aug[r][p] / aug[p][p];
        if (r != p) {
            for (int c = p; c < 128; ++c) aug[r][c] -= f * aug[p][c];
        }
        __syncthreads();
    }
    float d = aug[r][r];
    for (int c = 0; c < 64; ++c) Wc[r * 64 + c] = aug[r][64 + c] / d;
}

// -------- repack weight into WMMA B-operand register order (f16, K=64)
// B (K x N) packing used here, contiguous k order (same order as gemm64 A fragments):
//   lane 0-15 : column n = lane,   elements e=0..15 -> K = kstep*32 + e
//   lane 16-31: column n = lane-16, elements        -> K = kstep*32 + 16 + e
__global__ __launch_bounds__(256)
void repack_b_kernel(const float* __restrict__ W,  // !trans: [64, ncols]; trans: [ncols, 64]
                     _Float16* __restrict__ Bp,
                     int ncols, int trans)
{
    int idx = blockIdx.x * 256 + threadIdx.x;
    int total = (ncols >> 4) * 2 * 32 * 16;
    if (idx >= total) return;
    int e      = idx & 15;
    int lane   = (idx >> 4) & 31;
    int s      = (idx >> 9) & 1;
    int t      = idx >> 10;
    int lane16 = lane & 15;
    int hi     = lane >> 4;
    int k = s * 32 + hi * 16 + e;
    int n = t * 16 + lane16;
    float v = trans ? W[n * 64 + k] : W[k * ncols + n];
    Bp[idx] = (_Float16)v;
}

// -------------------------------------------------------- WMMA GEMM (K=64)
// out[m,n] = act( sum_k A[m,k]*B[k,n] + bias[n] + (HAS_ADD ? add[m,n] : 0) )
// One wave computes a 16 x NCOLS block.  A is f32, converted to f16 in-regs.
// ACT: 0 = none, 1 = relu, 2 = exact gelu
template <int NCOLS, int ACT, bool HAS_ADD>
__global__ __launch_bounds__(128)
void gemm64_kernel(const float* __restrict__ A,      // [M, 64]
                   const _Float16* __restrict__ Bp,  // packed
                   const float* __restrict__ bias,   // [NCOLS]
                   const float* __restrict__ add,    // [M, NCOLS] or null
                   float* __restrict__ out)          // [M, NCOLS]
{
    const int lane    = threadIdx.x & 31;
    const int waveId  = threadIdx.x >> 5;
    const int rowBase = (blockIdx.x * 4 + waveId) * 16;
    const int lane16  = lane & 15;
    const int hi      = lane >> 4;

    // A fragments in contiguous k order (matches packed B):
    // lanes 0-15: row M=lane, elems 0..15 -> K=s*32+0..15
    // lanes16-31: row M=lane-16, elems 0..15 -> K=s*32+16..31
    v16h afrag[2];
    const long long arow = (long long)(rowBase + lane16) * 64;
#pragma unroll
    for (int s = 0; s < 2; ++s) {
        const float* a0 = A + arow + s * 32 + hi * 16;
        const float* a1 = a0 + 8;
#pragma unroll
        for (int e = 0; e < 8; ++e) afrag[s][e] = (_Float16)a0[e];
#pragma unroll
        for (int e = 0; e < 8; ++e) afrag[s][8 + e] = (_Float16)a1[e];
    }

#pragma unroll
    for (int t = 0; t < (NCOLS >> 4); ++t) {
        v8f acc = {};
#pragma unroll
        for (int s = 0; s < 2; ++s) {
            v16h b = *(const v16h*)(Bp + ((t * 2 + s) * 32 + lane) * 16);
            acc = __builtin_amdgcn_wmma_f32_16x16x32_f16(
                false, afrag[s], false, b, (short)0, acc, false, false);
        }
        const int col = t * 16 + lane16;
        const float bv = bias ? bias[col] : 0.f;
#pragma unroll
        for (int i = 0; i < 8; ++i) {
            long long m = rowBase + hi * 8 + i;   // C/D layout: vgpr i, lane-half
            float v = acc[i] + bv;
            if (HAS_ADD) v += add[m * NCOLS + col];
            if (ACT == 1) v = fmaxf(v, 0.f);
            if (ACT == 2) v = 0.5f * v * (1.f + erff(v * 0.70710678118654752f));
            out[m * NCOLS + col] = v;
        }
    }
}

// ------------------------------------------- edge scatter: agg += z[src]*w
__global__ __launch_bounds__(256)
void edge_scatter_kernel(const int* __restrict__ src, const int* __restrict__ dst,
                         const float* __restrict__ ew,
                         const float* __restrict__ z,    // [N, 64]
                         float* __restrict__ agg)        // [N, 64]
{
    int tid = blockIdx.x * 256 + threadIdx.x;
    int e = tid >> 6;
    int c = tid & 63;
    int s = src[e];
    int d = dst[e];
    float w = ew[e];
    atomicAdd(&agg[(long long)d * 64 + c], z[(long long)s * 64 + c] * w);
}

// ------------------- Anderson Gram: gram[p] = <F_a - X_a, F_b - X_b>, a<=b
__global__ __launch_bounds__(256)
void gram_kernel(const float* __restrict__ Fb, const float* __restrict__ Xb,
                 long long stride, int n,
                 int s0, int s1, int s2, int s3, int s4,
                 float* __restrict__ gram, long long d)
{
    const int slots[5] = {s0, s1, s2, s3, s4};
    __shared__ float sg[15];
    if (threadIdx.x < 15) sg[threadIdx.x] = 0.f;
    __syncthreads();
    float acc[15];
#pragma unroll
    for (int p = 0; p < 15; ++p) acc[p] = 0.f;
    long long i  = (long long)blockIdx.x * 256 + threadIdx.x;
    long long st = (long long)gridDim.x * 256;
    for (; i < d; i += st) {
        float g[5];
#pragma unroll
        for (int a = 0; a < 5; ++a)
            g[a] = (a < n) ? (Fb[slots[a] * stride + i] - Xb[slots[a] * stride + i]) : 0.f;
        int p = 0;
#pragma unroll
        for (int a = 0; a < 5; ++a)
#pragma unroll
            for (int b = a; b < 5; ++b) { acc[p] += g[a] * g[b]; ++p; }
    }
#pragma unroll
    for (int p = 0; p < 15; ++p) atomicAdd(&sg[p], acc[p]);
    __syncthreads();
    if (threadIdx.x < 15) atomicAdd(&gram[threadIdx.x], sg[threadIdx.x]);
}

// ------------------------- tiny solve: (G + lam I) a = 1 ; a /= sum(a)
__global__ void solve_kernel(const float* __restrict__ gram,
                             float* __restrict__ alpha, int n, float lam)
{
    if (threadIdx.x != 0 || blockIdx.x != 0) return;
    float full[5][5];
    int p = 0;
    for (int a = 0; a < 5; ++a)
        for (int b = a; b < 5; ++b) { full[a][b] = gram[p]; full[b][a] = gram[p]; ++p; }
    float Hm[5][6];
    for (int i = 0; i < n; ++i) {
        for (int j = 0; j < n; ++j) Hm[i][j] = full[i][j] + ((i == j) ? lam : 0.f);
        Hm[i][n] = 1.f;
    }
    for (int c = 0; c < n; ++c) {
        int piv = c;
        for (int r = c + 1; r < n; ++r)
            if (fabsf(Hm[r][c]) > fabsf(Hm[piv][c])) piv = r;
        if (piv != c)
            for (int j = 0; j <= n; ++j) { float t = Hm[c][j]; Hm[c][j] = Hm[piv][j]; Hm[piv][j] = t; }
        float inv = 1.f / Hm[c][c];
        for (int j = c; j <= n; ++j) Hm[c][j] *= inv;
        for (int r = 0; r < n; ++r) {
            if (r == c) continue;
            float f = Hm[r][c];
            for (int j = c; j <= n; ++j) Hm[r][j] -= f * Hm[c][j];
        }
    }
    float s = 0.f;
    for (int i = 0; i < n; ++i) s += Hm[i][n];
    for (int i = 0; i < 5; ++i) alpha[i] = (i < n) ? Hm[i][n] / s : 0.f;
}

// -------------------------------------- xnew = sum_i alpha[i] * F_slot[i]
__global__ __launch_bounds__(256)
void combine_kernel(const float* __restrict__ Fb, long long stride,
                    const float* __restrict__ alpha, int n,
                    int s0, int s1, int s2, int s3, int s4,
                    float* __restrict__ xnew, long long d)
{
    const int slots[5] = {s0, s1, s2, s3, s4};
    float al[5];
#pragma unroll
    for (int a = 0; a < 5; ++a) al[a] = alpha[a];
    long long i  = (long long)blockIdx.x * 256 + threadIdx.x;
    long long st = (long long)gridDim.x * 256;
    for (; i < d; i += st) {
        float v = 0.f;
#pragma unroll
        for (int a = 0; a < 5; ++a)
            if (a < n) v += al[a] * Fb[slots[a] * stride + i];
        xnew[i] = v;
    }
}

// ---------------------------------------- batchnorm stats (sum, sumsq)
__global__ __launch_bounds__(256)
void bn_stats_kernel(const float* __restrict__ x, float* __restrict__ sums)
{
    __shared__ float ssum[64], ssq[64];
    if (threadIdx.x < 64) { ssum[threadIdx.x] = 0.f; ssq[threadIdx.x] = 0.f; }
    __syncthreads();
    int c = threadIdx.x & 63;   // stride 256 == 0 mod 64 -> channel constant
    float ls = 0.f, lq = 0.f;
    long long i  = (long long)blockIdx.x * 256 + threadIdx.x;
    long long st = (long long)gridDim.x * 256;
    for (; i < kD; i += st) {
        float v = x[i];
        ls += v; lq += v * v;
    }
    atomicAdd(&ssum[c], ls);
    atomicAdd(&ssq[c], lq);
    __syncthreads();
    if (threadIdx.x < 64) {
        atomicAdd(&sums[threadIdx.x], ssum[threadIdx.x]);
        atomicAdd(&sums[64 + threadIdx.x], ssq[threadIdx.x]);
    }
}

// -------------------------- BN apply + relu + segment-sum pooling (fused)
__global__ __launch_bounds__(256)
void bn_pool_kernel(const float* __restrict__ x, const float* __restrict__ sums,
                    const float* __restrict__ gamma, const float* __restrict__ beta,
                    const int* __restrict__ batch,
                    float* __restrict__ pooled, float* __restrict__ cnt)
{
    int tid  = blockIdx.x * 256 + threadIdx.x;
    int node = tid >> 6;
    int c    = tid & 63;
    float mu  = sums[c] * (1.f / kN);
    float var = sums[64 + c] * (1.f / kN) - mu * mu;
    float v = x[(long long)node * 64 + c];
    v = gamma[c] * (v - mu) * rsqrtf(var + 1e-5f) + beta[c];
    v = fmaxf(v, 0.f);
    int g = batch[node];
    atomicAdd(&pooled[g * 64 + c], v);
    if (c == 0) atomicAdd(&cnt[g], 1.f);
}

__global__ __launch_bounds__(256)
void pool_div_kernel(float* __restrict__ pooled, const float* __restrict__ cnt)
{
    int tid = blockIdx.x * 256 + threadIdx.x;
    if (tid >= kG * 64) return;
    pooled[tid] /= fmaxf(cnt[tid >> 6], 1.f);
}

} // anonymous namespace

extern "C" void kernel_launch(void* const* d_in, const int* in_sizes, int n_in,
                              void* d_out, int out_size, void* d_ws, size_t ws_size,
                              hipStream_t stream)
{
    (void)in_sizes; (void)n_in; (void)out_size; (void)ws_size;

    const int*   data_x = (const int*)  d_in[0];
    const int*   eidx   = (const int*)  d_in[1];
    const float* ew     = (const float*)d_in[2];
    const int*   batch  = (const int*)  d_in[3];
    const float* emb    = (const float*)d_in[4];
    const float* W1     = (const float*)d_in[5];
    const float* b1     = (const float*)d_in[6];
    const float* W2     = (const float*)d_in[7];
    const float* b2     = (const float*)d_in[8];
    const float* Mw     = (const float*)d_in[9];
    const float* cb     = (const float*)d_in[10];
    const float* gamma  = (const float*)d_in[11];
    const float* beta   = (const float*)d_in[12];
    const float* W3     = (const float*)d_in[13];
    const float* b3     = (const float*)d_in[14];
    const float* W4     = (const float*)d_in[15];
    const float* b4     = (const float*)d_in[16];
    float* out = (float*)d_out;

    const int* esrc = eidx;          // edge_index[0]
    const int* edst = eidx + kE;     // edge_index[1]

    // ----------------------------- workspace layout (floats) ------------
    float* ws     = (float*)d_ws;
    float* X      = ws;                   // 5 * kD  (Anderson X history)
    float* F      = X + 5 * kD;           // 5 * kD  (Anderson F history)
    float* agg    = F + 5 * kD;           // kD
    float* xinj   = agg + kD;             // kD
    float* tmp    = xinj + kD;            // kD
    float* Wc     = tmp + kD;             // 4096
    float* stats  = Wc + 4096;            // 128 (sum | sumsq)
    float* pooled = stats + 128;          // kG*64
    float* cnt    = pooled + kG * 64;     // kG
    float* gram   = cnt + kG;             // 16
    float* alpha  = gram + 16;            // 8
    float* decT   = alpha + 8;            // kG*64
    _Float16* W1p = (_Float16*)(decT + kG * 64);  // 4096 halves each
    _Float16* W2p = W1p + 4096;
    _Float16* W3p = W2p + 4096;
    _Float16* Wcp = W3p + 4096;
    _Float16* W4p = Wcp + 4096;           // 8192 halves

    // ----------------------------- weight prep --------------------------
    repack_b_kernel<<<16, 256, 0, stream>>>(W1, W1p, 64, 0);
    repack_b_kernel<<<16, 256, 0, stream>>>(W2, W2p, 64, 0);
    repack_b_kernel<<<16, 256, 0, stream>>>(W3, W3p, 64, 0);
    repack_b_kernel<<<32, 256, 0, stream>>>(W4, W4p, 128, 0);
    cayley_kernel<<<1, 64, 0, stream>>>(Mw, Wc);
    repack_b_kernel<<<16, 256, 0, stream>>>(Wc, Wcp, 64, 1);  // B = Wc^T

    // ----------------------------- encode + enc MLP ---------------------
    atom_enc_kernel<<<kN * kH / 256, 256, 0, stream>>>(data_x, emb, tmp);
    gemm64_kernel<64, 2, false><<<kN / 64, 128, 0, stream>>>(tmp, W1p, b1, nullptr, agg);
    gemm64_kernel<64, 0, false><<<kN / 64, 128, 0, stream>>>(agg, W2p, b2, nullptr, xinj);

    // f(z) = relu( (z.A) @ Wc^T + cb + x_inj ), node-major [N,64]
    auto feval = [&](const float* z, float* fout) {
        fill_kernel<<<2048, 256, 0, stream>>>(agg, 0.f, kD);
        edge_scatter_kernel<<<kE * kH / 256, 256, 0, stream>>>(esrc, edst, ew, z, agg);
        gemm64_kernel<64, 1, true><<<kN / 64, 128, 0, stream>>>(agg, Wcp, cb, xinj, fout);
    };

    // ----------------------------- Anderson -----------------------------
    fill_kernel<<<2048, 256, 0, stream>>>(X, 0.f, kD);           // X[0] = 0
    feval(X, F);                                                 // F[0] = f(x0)
    hipMemcpyAsync(X + kD, F, kD * sizeof(float),
                   hipMemcpyDeviceToDevice, stream);             // X[1] = f0
    feval(X + kD, F + kD);                                       // F[1] = f(f0)

    for (int k = 2; k < 10; ++k) {
        int n = (k < 5) ? k : 5;
        int w[5] = {0, 0, 0, 0, 0};
        for (int i = 0; i < n; ++i) w[i] = (k - n + i) % 5;      // circular slots
        fill_kernel<<<1, 32, 0, stream>>>(gram, 0.f, 16);
        gram_kernel<<<512, 256, 0, stream>>>(F, X, kD, n,
                                             w[0], w[1], w[2], w[3], w[4], gram, kD);
        solve_kernel<<<1, 1, 0, stream>>>(gram, alpha, n, 1e-4f);
        combine_kernel<<<1024, 256, 0, stream>>>(F, kD, alpha, n,
                                                 w[0], w[1], w[2], w[3], w[4],
                                                 X + (long long)(k % 5) * kD, kD);
        feval(X + (long long)(k % 5) * kD, F + (long long)(k % 5) * kD);
    }
    const float* zstar = F + (long long)(9 % 5) * kD;            // Fh[-1]

    // ----------------------------- BN + pool + dec ----------------------
    fill_kernel<<<1, 128, 0, stream>>>(stats, 0.f, 128);
    bn_stats_kernel<<<1024, 256, 0, stream>>>(zstar, stats);
    fill_kernel<<<64, 256, 0, stream>>>(pooled, 0.f, kG * 64);
    fill_kernel<<<1, 256, 0, stream>>>(cnt, 0.f, kG);
    bn_pool_kernel<<<kN * kH / 256, 256, 0, stream>>>(zstar, stats, gamma, beta,
                                                      batch, pooled, cnt);
    pool_div_kernel<<<kG * 64 / 256, 256, 0, stream>>>(pooled, cnt);
    gemm64_kernel<64, 2, false><<<kG / 64, 128, 0, stream>>>(pooled, W3p, b3, nullptr, decT);
    gemm64_kernel<128, 0, false><<<kG / 64, 128, 0, stream>>>(decT, W4p, b4, nullptr, out);
}
